// EKMFKT_7954279432879
// MI455X (gfx1250) — compile-verified, matrix operand order checked by reading, not run on hardware
//
#include <hip/hip_runtime.h>
#include <hip/hip_bf16.h>

// EKMFKT forward for MI455X (gfx1250, wave32).
// All GEMMs via V_WMMA_F32_16X16X32_BF16 (bf16 in, fp32 accumulate).
// Scan kernel keeps both gate weight matrices resident in LDS (282KB of the
// 320KB/WGP) so every WMMA operand is ds-fed; ci/cb staging is software-
// pipelined one step ahead through registers.

typedef __attribute__((ext_vector_type(16))) __bf16 v16bf;
typedef __attribute__((ext_vector_type(8)))  float  v8f;

#define BB 256
#define SS 512
#define DD 128
#define HH 128

__device__ __forceinline__ __bf16 f2bf(float f) {
  unsigned u = __builtin_bit_cast(unsigned, f);
  unsigned r = u + 0x7FFFu + ((u >> 16) & 1u);   // round-to-nearest-even
  unsigned short h = (unsigned short)(r >> 16);
  return __builtin_bit_cast(__bf16, h);
}
__device__ __forceinline__ float bf2f(__bf16 x) {
  unsigned short h = __builtin_bit_cast(unsigned short, x);
  unsigned u = ((unsigned)h) << 16;
  return __builtin_bit_cast(float, u);
}
__device__ __forceinline__ float sigmoidf_(float x) {
  return 1.0f / (1.0f + __expf(-x));
}

// Swizzle for 16x16x32 bf16 WMMA operand tiles:
// element index within a [kiter][lane][16] block for (row_or_col rc, k).
__device__ __forceinline__ int bswz(int rc, int k) {
  int kiter = k >> 5;
  int kl = k & 31;
  int half, e;
  if (kl < 16) { half = kl >> 3; e = kl & 7; }
  else         { half = (kl - 16) >> 3; e = 8 + ((kl - 16) & 7); }
  int lane = (half << 4) | rc;
  return (kiter * 32 + lane) * 16 + e;
}

__device__ __forceinline__ v8f wmma_bf16(const __bf16* a_p, const __bf16* b_p, v8f c) {
  v16bf a = *reinterpret_cast<const v16bf*>(a_p);
  v16bf b = *reinterpret_cast<const v16bf*>(b_p);
  return __builtin_amdgcn_wmma_f32_16x16x32_bf16(false, a, false, b, (short)0, c,
                                                 false, false);
}

// ---------------------------------------------------------------------------
// Kernel 1: weight prep — bf16 convert + swizzle into WMMA B layout.
// W_learn (513 rows) reordered to [h, ci, nb, cb] (512 rows); iv row stored
// separately (folded into accumulator init in the scan).
// ---------------------------------------------------------------------------
__global__ void ekmfkt_prep(const float* __restrict__ W_int,
                            const float* __restrict__ W_learn,
                            const float* __restrict__ W_forget,
                            const float* __restrict__ W_b2,
                            __bf16* __restrict__ Wint_sw,
                            __bf16* __restrict__ Wl_sw,
                            __bf16* __restrict__ Wf_sw,
                            __bf16* __restrict__ Wb2_sw,
                            float* __restrict__ Wiv) {
  int idx = blockIdx.x * blockDim.x + threadIdx.x;
  const int NK = 512 * 128;
  const int NT512 = 16 * 32 * 16;  // elems per n-tile, K=512
  const int NT128 = 4 * 32 * 16;   // elems per n-tile, K=128
  if (idx < NK) {                  // W_int [512,128]
    int k = idx >> 7, n = idx & 127;
    Wint_sw[(n >> 4) * NT512 + bswz(n & 15, k)] = f2bf(W_int[k * 128 + n]);
    return;
  }
  idx -= NK;
  if (idx < NK) {                  // W_forget [512,128] — concat order matches
    int k = idx >> 7, n = idx & 127;
    Wf_sw[(n >> 4) * NT512 + bswz(n & 15, k)] = f2bf(W_forget[k * 128 + n]);
    return;
  }
  idx -= NK;
  if (idx < NK) {                  // W_learn reordered [h,ci,nb,cb], drop iv row
    int k = idx >> 7, n = idx & 127;
    int src = (k < 256) ? k : ((k < 384) ? (385 + (k - 256)) : (257 + (k - 384)));
    Wl_sw[(n >> 4) * NT512 + bswz(n & 15, k)] = f2bf(W_learn[src * 128 + n]);
    return;
  }
  idx -= NK;
  if (idx < 128 * 128) {           // W_b2 [128,128]
    int k = idx >> 7, n = idx & 127;
    Wb2_sw[(n >> 4) * NT128 + bswz(n & 15, k)] = f2bf(W_b2[k * 128 + n]);
    return;
  }
  idx -= 128 * 128;
  if (idx < 128) Wiv[idx] = W_learn[256 * 128 + idx];  // iv row of W_learn
}

// ---------------------------------------------------------------------------
// Kernel 2: parallel precompute of ci = relu([e,s,t,r]@W_int+b) and
// cb = relu(beh_in@W_b1+b1)@W_b2+b2, stored bf16. 16 rows/block, 8 waves,
// each wave owns one 16-wide N-tile.
// ---------------------------------------------------------------------------
__global__ __launch_bounds__(256)
void ekmfkt_precompute(const int* __restrict__ exercise_seq,
                       const int* __restrict__ skill_seq,
                       const int* __restrict__ response_seq,
                       const float* __restrict__ time_seq,
                       const float* __restrict__ attempt_seq,
                       const float* __restrict__ hint_seq,
                       const float* __restrict__ exercise_emb,
                       const float* __restrict__ skill_emb,
                       const float* __restrict__ response_emb,
                       const float* __restrict__ W_time,
                       const float* __restrict__ b_time,
                       const float* __restrict__ b_int,
                       const float* __restrict__ W_b1,
                       const float* __restrict__ b_b1,
                       const float* __restrict__ b_b2,
                       const __bf16* __restrict__ Wint_sw,
                       const __bf16* __restrict__ Wb2_sw,
                       __bf16* __restrict__ ci_bf,
                       __bf16* __restrict__ cb_bf) {
  __shared__ __align__(32) __bf16 Asw[16 * 32 * 16];   // 16x512 swizzled
  __shared__ __align__(32) __bf16 Asw2[4 * 32 * 16];   // 16x128 swizzled
  __shared__ int   eix[16], six[16], rix[16];
  __shared__ float tv[16], av[16], hv[16];

  const int tid = threadIdx.x;
  const int g0 = blockIdx.x * 16;   // first (b,s) flat row
  if (tid < 16) {
    int g = g0 + tid;
    eix[tid] = exercise_seq[g];
    six[tid] = skill_seq[g];
    rix[tid] = response_seq[g];
    tv[tid]  = time_seq[g];
    av[tid]  = attempt_seq[g];
    hv[tid]  = hint_seq[g];
  }
  __syncthreads();

  // stage x = [e, s, t, r] (16x512) into WMMA A layout
  for (int idx = tid; idx < 16 * 512; idx += 256) {
    int r = idx >> 9, k = idx & 511;
    float v;
    if (k < 128)      v = exercise_emb[(size_t)eix[r] * 128 + k];
    else if (k < 256) v = skill_emb[(size_t)six[r] * 128 + (k - 128)];
    else if (k < 384) v = tv[r] * W_time[k - 256] + b_time[k - 256];
    else              v = response_emb[(size_t)rix[r] * 128 + (k - 384)];
    Asw[bswz(r, k)] = f2bf(v);
  }
  // stage hidden = relu(beh_in@W_b1+b_b1) (16x128)
  for (int idx = tid; idx < 16 * 128; idx += 256) {
    int r = idx >> 7, j = idx & 127;
    float h = av[r] * W_b1[j] + hv[r] * W_b1[128 + j] + b_b1[j];
    Asw2[bswz(r, j)] = f2bf(fmaxf(h, 0.0f));
  }
  __syncthreads();

  const int wave = tid >> 5, lane = tid & 31;
  const int col = wave * 16 + (lane & 15);
  const int rbase = (lane >> 4) * 8;

  {  // inter GEMM: K=512, 16 k-iters, 2 accumulator chains
    v8f c0, c1;
    float bi = b_int[col];
#pragma unroll
    for (int v = 0; v < 8; ++v) { c0[v] = bi; c1[v] = 0.0f; }
    const __bf16* Wt = Wint_sw + wave * (16 * 32 * 16);
#pragma unroll
    for (int kit = 0; kit < 16; kit += 2) {
      c0 = wmma_bf16(&Asw[(kit * 32 + lane) * 16], &Wt[(kit * 32 + lane) * 16], c0);
      c1 = wmma_bf16(&Asw[((kit + 1) * 32 + lane) * 16],
                     &Wt[((kit + 1) * 32 + lane) * 16], c1);
    }
#pragma unroll
    for (int v = 0; v < 8; ++v)
      ci_bf[(size_t)(g0 + rbase + v) * 128 + col] = f2bf(fmaxf(c0[v] + c1[v], 0.0f));
  }
  {  // beh GEMM: K=128, 4 k-iters (no relu on output)
    v8f c0, c1;
    float bb = b_b2[col];
#pragma unroll
    for (int v = 0; v < 8; ++v) { c0[v] = bb; c1[v] = 0.0f; }
    const __bf16* Wt = Wb2_sw + wave * (4 * 32 * 16);
#pragma unroll
    for (int kit = 0; kit < 4; kit += 2) {
      c0 = wmma_bf16(&Asw2[(kit * 32 + lane) * 16], &Wt[(kit * 32 + lane) * 16], c0);
      c1 = wmma_bf16(&Asw2[((kit + 1) * 32 + lane) * 16],
                     &Wt[((kit + 1) * 32 + lane) * 16], c1);
    }
#pragma unroll
    for (int v = 0; v < 8; ++v)
      cb_bf[(size_t)(g0 + rbase + v) * 128 + col] = f2bf(c0[v] + c1[v]);
  }
}

// ---------------------------------------------------------------------------
// Kernel 3: sequential scan. 16 blocks x 16 batch rows; 8 waves, each owns a
// 16-col N-tile of h. Wl/Wf (256KB) resident in LDS; A-tile + fp32 h in LDS;
// ci/cb prefetched one step ahead in registers. 32 WMMA per wave per step.
// ---------------------------------------------------------------------------
__global__ __launch_bounds__(256)
void ekmfkt_scan(const __bf16* __restrict__ ci_bf,
                 const __bf16* __restrict__ cb_bf,
                 const __bf16* __restrict__ Wl_g,
                 const __bf16* __restrict__ Wf_g,
                 const float* __restrict__ Wiv,
                 const float* __restrict__ b_learn,
                 const float* __restrict__ b_forget,
                 const float* __restrict__ interval_seq,
                 const int* __restrict__ exercise_seq,
                 const float* __restrict__ exercise_emb,
                 const float* __restrict__ W_pred,
                 const float* __restrict__ b_pred,
                 float* __restrict__ out) {
  extern __shared__ __align__(32) char smem[];
  __bf16* Wl    = (__bf16*)smem;            // 65536 elems = 128 KB
  __bf16* Wf    = Wl + 512 * 128;           // 128 KB
  __bf16* Asw   = Wf + 512 * 128;           // 8192 elems = 16 KB
  float*  h_lds = (float*)(Asw + 8192);     // 8 KB
  float*  red   = h_lds + 16 * 128;         // 1 KB
  float*  ivl   = red + 256;                // 64 B
  int*    eixl  = (int*)(ivl + 16);         // 64 B

  const int tid = threadIdx.x;
  const int bbase = blockIdx.x * 16;

  // one-time copy of pre-swizzled weights into LDS (uint4 = 8 bf16)
  {
    const uint4* s0 = (const uint4*)Wl_g;
    const uint4* s1 = (const uint4*)Wf_g;
    uint4* d0 = (uint4*)Wl;
    uint4* d1 = (uint4*)Wf;
    for (int i = tid; i < 512 * 128 / 8; i += 256) { d0[i] = s0[i]; d1[i] = s1[i]; }
  }
  for (int i = tid; i < 16 * 128; i += 256) h_lds[i] = 0.0f;

  const int wave = tid >> 5, lane = tid & 31;
  const int col = wave * 16 + (lane & 15);
  const int rbase = (lane >> 4) * 8;
  const float bl = b_learn[col];
  const float bfg = b_forget[col];
  const float wiv = Wiv[col];
  const __bf16* WlT = Wl + wave * (16 * 32 * 16);
  const __bf16* WfT = Wf + wave * (16 * 32 * 16);
  const int prow = tid >> 4, pseg = tid & 15;
  const int srow = tid >> 4;                 // staging: row owned by thread
  const int sch  = tid & 15;                 // staging: 8-elem chunk
  const size_t srow_seq = (size_t)(bbase + srow) * SS;

  // prefetch t=0 activations (8 bf16 per tensor per thread)
  uint4 ci_pre = *(const uint4*)(ci_bf + srow_seq * 128 + sch * 8);
  uint4 cb_pre = *(const uint4*)(cb_bf + srow_seq * 128 + sch * 8);
  __syncthreads();

  for (int t = 0; t < SS; ++t) {
    // stage A = [h | ci | nb | cb] (nb == cb except last step)
    {
      const __bf16* cip = (const __bf16*)&ci_pre;
      const __bf16* cbp = (const __bf16*)&cb_pre;
      const bool last = (t == SS - 1);
      const __bf16 z = f2bf(0.0f);
#pragma unroll
      for (int j = 0; j < 8; ++j) {
        int kk = sch * 8 + j;
        Asw[bswz(srow, kk)]       = f2bf(h_lds[srow * 128 + kk]);
        Asw[bswz(srow, 128 + kk)] = cip[j];
        Asw[bswz(srow, 256 + kk)] = last ? z : cbp[j];
        Asw[bswz(srow, 384 + kk)] = cbp[j];
      }
    }
    if (tid < 16) {
      int g = (bbase + tid) * SS + t;
      ivl[tid]  = interval_seq[g];
      eixl[tid] = exercise_seq[g];
    }
    __syncthreads();

    // prefetch step t+1 — latency hides behind the two gate GEMMs
    if (t + 1 < SS) {
      ci_pre = *(const uint4*)(ci_bf + (srow_seq + t + 1) * 128 + sch * 8);
      cb_pre = *(const uint4*)(cb_bf + (srow_seq + t + 1) * 128 + sch * 8);
    }

    // learn gate: C init = b_learn + iv * W_learn[iv_row]; 2 accum chains
    v8f c0, c1;
#pragma unroll
    for (int v = 0; v < 8; ++v) { c0[v] = bl + ivl[rbase + v] * wiv; c1[v] = 0.0f; }
#pragma unroll
    for (int kit = 0; kit < 16; kit += 2) {
      c0 = wmma_bf16(&Asw[(kit * 32 + lane) * 16], &WlT[(kit * 32 + lane) * 16], c0);
      c1 = wmma_bf16(&Asw[((kit + 1) * 32 + lane) * 16],
                     &WlT[((kit + 1) * 32 + lane) * 16], c1);
    }
    v8f lg;
#pragma unroll
    for (int v = 0; v < 8; ++v) lg[v] = sigmoidf_(c0[v] + c1[v]);
    __syncthreads();  // all waves done reading nb segment

    // overwrite segment X with lg for the forget gate
#pragma unroll
    for (int v = 0; v < 8; ++v)
      Asw[bswz(rbase + v, 256 + col)] = f2bf(lg[v]);
    __syncthreads();

    // forget gate
#pragma unroll
    for (int v = 0; v < 8; ++v) { c0[v] = bfg; c1[v] = 0.0f; }
#pragma unroll
    for (int kit = 0; kit < 16; kit += 2) {
      c0 = wmma_bf16(&Asw[(kit * 32 + lane) * 16], &WfT[(kit * 32 + lane) * 16], c0);
      c1 = wmma_bf16(&Asw[((kit + 1) * 32 + lane) * 16],
                     &WfT[((kit + 1) * 32 + lane) * 16], c1);
    }
    float hn[8];
#pragma unroll
    for (int v = 0; v < 8; ++v) {
      float f = sigmoidf_(c0[v] + c1[v]);
      hn[v] = lg[v] + f * h_lds[(rbase + v) * 128 + col];
    }
#pragma unroll
    for (int v = 0; v < 8; ++v) h_lds[(rbase + v) * 128 + col] = hn[v];
    __syncthreads();

    // prediction: sigmoid([h, e] @ W_pred + b_pred), 16 threads per row
    float ps = 0.0f;
#pragma unroll
    for (int j = 0; j < 16; ++j) {
      int idx = pseg * 16 + j;
      float x = (idx < 128) ? h_lds[prow * 128 + idx]
                            : exercise_emb[(size_t)eixl[prow] * 128 + (idx - 128)];
      ps += x * W_pred[idx];
    }
    red[prow * 16 + pseg] = ps;
    __syncthreads();
    if (pseg == 0) {
      float s = b_pred[0];
#pragma unroll
      for (int j = 0; j < 16; ++j) s += red[prow * 16 + j];
      out[(size_t)(bbase + prow) * SS + t] = sigmoidf_(s);
    }
    __syncthreads();
  }
}

// ---------------------------------------------------------------------------
extern "C" void kernel_launch(void* const* d_in, const int* in_sizes, int n_in,
                              void* d_out, int out_size, void* d_ws, size_t ws_size,
                              hipStream_t stream) {
  const int*   exercise_seq = (const int*)d_in[0];
  const int*   skill_seq    = (const int*)d_in[1];
  const int*   response_seq = (const int*)d_in[2];
  const float* time_seq     = (const float*)d_in[3];
  const float* interval_seq = (const float*)d_in[4];
  const float* attempt_seq  = (const float*)d_in[5];
  const float* hint_seq     = (const float*)d_in[6];
  // d_in[7] = q_matrix (unused by the reference forward)
  const float* exercise_emb = (const float*)d_in[8];
  const float* skill_emb    = (const float*)d_in[9];
  const float* response_emb = (const float*)d_in[10];
  const float* W_time  = (const float*)d_in[11];
  const float* b_time  = (const float*)d_in[12];
  const float* W_int   = (const float*)d_in[13];
  const float* b_int   = (const float*)d_in[14];
  const float* W_b1    = (const float*)d_in[15];
  const float* b_b1    = (const float*)d_in[16];
  const float* W_b2    = (const float*)d_in[17];
  const float* b_b2    = (const float*)d_in[18];
  const float* W_learn = (const float*)d_in[19];
  const float* b_learn = (const float*)d_in[20];
  const float* W_forget= (const float*)d_in[21];
  const float* b_forget= (const float*)d_in[22];
  const float* W_pred  = (const float*)d_in[23];
  const float* b_pred  = (const float*)d_in[24];

  // workspace layout (bytes); total ~67.6 MB
  char* ws = (char*)d_ws;
  __bf16* ci_bf   = (__bf16*)(ws + 0);          // 256*512*128 bf16 = 32 MB
  __bf16* cb_bf   = (__bf16*)(ws + 33554432);   // 32 MB
  __bf16* Wl_sw   = (__bf16*)(ws + 67108864);   // 128 KB
  __bf16* Wf_sw   = (__bf16*)(ws + 67239936);   // 128 KB
  __bf16* Wint_sw = (__bf16*)(ws + 67371008);   // 128 KB
  __bf16* Wb2_sw  = (__bf16*)(ws + 67502080);   // 32 KB
  float*  Wiv     = (float*) (ws + 67534848);   // 512 B

  const int prep_elems = 3 * 512 * 128 + 128 * 128 + 128;
  ekmfkt_prep<<<(prep_elems + 255) / 256, 256, 0, stream>>>(
      W_int, W_learn, W_forget, W_b2, Wint_sw, Wl_sw, Wf_sw, Wb2_sw, Wiv);

  ekmfkt_precompute<<<(BB * SS) / 16, 256, 0, stream>>>(
      exercise_seq, skill_seq, response_seq, time_seq, attempt_seq, hint_seq,
      exercise_emb, skill_emb, response_emb, W_time, b_time, b_int,
      W_b1, b_b1, b_b2, Wint_sw, Wb2_sw, ci_bf, cb_bf);

  // dynamic LDS: 2x128KB weights + 16KB A-tile + 8KB h + ~1.2KB misc = 287872 B
  // (fits gfx1250's 320KB-per-workgroup LDS)
  const size_t scan_lds = 2 * 512 * 128 * sizeof(__bf16)   // Wl + Wf
                        + 8192 * sizeof(__bf16)            // Asw
                        + 16 * 128 * sizeof(float)         // h_lds
                        + 256 * sizeof(float)              // red
                        + 16 * sizeof(float)               // ivl
                        + 16 * sizeof(int);                // eixl
  ekmfkt_scan<<<BB / 16, 256, scan_lds, stream>>>(
      ci_bf, cb_bf, Wl_sw, Wf_sw, Wiv, b_learn, b_forget,
      interval_seq, exercise_seq, exercise_emb, W_pred, b_pred,
      (float*)d_out);
}
